// LSTM_41583873360224
// MI455X (gfx1250) — matrix operand order repaired and checked by reading, not compiled, not run on hardware
//
#include <hip/hip_runtime.h>
#include <stdint.h>

#define BATCH   256
#define TSTEPS  1024
#define HID     256
#define NCLS    10
#define NWG     8            // workgroups; each owns HID/NWG = 32 output columns
#define NCOL    32
#define THREADS 256          // 8 wave32
#define LDH     264          // padded bf16 row length for h in LDS (+16B -> conflict-free)
#define LDW     264          // padded bf16 K length per weight column in LDS

typedef __attribute__((ext_vector_type(16))) __bf16       v16bf;
typedef __attribute__((ext_vector_type(8)))  __bf16       v8bf;
typedef __attribute__((ext_vector_type(8)))  float        v8f;
typedef __attribute__((ext_vector_type(4)))  unsigned int v4u;
typedef __attribute__((ext_vector_type(8)))  int          v8i;
typedef __attribute__((ext_vector_type(4)))  int          v4i;

struct SMem {
  __bf16 h[BATCH * LDH];          // MUST be first: TDM writes LDS offset 0
  __bf16 w[4][NCOL][LDW];         // [gate][local col][K] (K-contiguous)
};

__device__ __forceinline__ __bf16 to_bf16(float f) {
  unsigned u = __builtin_bit_cast(unsigned, f);
  u += 0x7fffu + ((u >> 16) & 1u);                 // round-to-nearest-even
  unsigned short s = (unsigned short)(u >> 16);
  return __builtin_bit_cast(__bf16, s);
}
__device__ __forceinline__ float fast_sig(float x)  { return 1.f / (1.f + __expf(-x)); }
__device__ __forceinline__ float fast_tanh(float x) { float e = __expf(2.f * x); return (e - 1.f) / (e + 1.f); }

__device__ __forceinline__ void grid_barrier(unsigned* cnt, unsigned target) {
  __threadfence();                 // each thread's stores reach device scope
  __syncthreads();
  if (threadIdx.x == 0) {
    __hip_atomic_fetch_add(cnt, 1u, __ATOMIC_RELEASE, __HIP_MEMORY_SCOPE_AGENT);
    while (__hip_atomic_load(cnt, __ATOMIC_ACQUIRE, __HIP_MEMORY_SCOPE_AGENT) < target)
      __builtin_amdgcn_s_sleep(2);
  }
  __syncthreads();
}

__global__ __launch_bounds__(THREADS)
void lstm_persistent(const float* __restrict__ x,
                     const float* wgx, const float* wgh, const float* bg,
                     const float* wix, const float* wih, const float* bi,
                     const float* wfx, const float* wfh, const float* bfv,
                     const float* wox, const float* woh, const float* bo,
                     const float* wph, const float* bp, const float* hinit,
                     __bf16* hg, float* h32, unsigned* cnt, float* out)
{
  __shared__ SMem sm;
  const int tid   = threadIdx.x;
  const int wave  = tid >> 5;
  const int lane  = tid & 31;
  const int ntile = wave & 1;          // which 16-col tile of the WG's 32 cols
  const int mgrp  = wave >> 1;         // m-tiles {mgrp, mgrp+4, mgrp+8, mgrp+12}
  const int n0    = blockIdx.x * NCOL;
  const int colg  = n0 + ntile * 16 + (lane & 15);
  const int lhi   = lane >> 4;

  const float* Wh[4] = {wgh, wih, wfh, woh};
  const float* Wx[4] = {wgx, wix, wfx, wox};
  const float* Bb[4] = {bg,  bi,  bfv, bo };

  // ---- one-time: stage this WG's recurrent weights into LDS as bf16 ----
  for (int i = tid; i < 4 * NCOL * HID; i += THREADS) {
    int g  = i >> 13;          // / (NCOL*HID)
    int r  = i & 8191;
    int nc = r >> 8;           // / HID
    int k  = r & 255;
    sm.w[g][nc][k] = to_bf16(Wh[g][k * HID + n0 + nc]);
  }
  // ---- one-time: initial h slice (broadcast h_init over batch) ----
  for (int j = 0; j < NCOL; ++j)
    hg[tid * HID + n0 + j] = to_bf16(hinit[n0 + j]);

  float wxv[4], bsv[4];
  #pragma unroll
  for (int g = 0; g < 4; ++g) { wxv[g] = Wx[g][colg]; bsv[g] = Bb[g][colg]; }

  float cst[4][8];                      // cell state, register-resident for all T steps
  #pragma unroll
  for (int j = 0; j < 4; ++j)
    #pragma unroll
    for (int r = 0; r < 8; ++r) cst[j][r] = 0.f;

  unsigned bar = 0;
  grid_barrier(cnt, ++bar * NWG);       // all WGs wrote their h0 slice

  for (int t = 0; t < TSTEPS; ++t) {
    // ---- TDM: broadcast full h [256x256] bf16 -> LDS, row-padded to LDH ----
    if (wave == 0) {
      unsigned long long ga = (unsigned long long)(uintptr_t)hg;
      v4u g0 = { 1u,                                   // count=1, user descriptor
                 0u,                                   // lds_addr = 0 (sm.h)
                 (unsigned)ga,
                 (unsigned)((ga >> 32) & 0x01ffffffu) | (2u << 30) }; // type=2
      v8i g1 = { (int)((1u << 16)        // data_size = 1 (2 bytes)
                     | (1u << 20)        // pad_enable
                     | (6u << 22)        // pad_interval: 128 dwords (one 512B row)
                     | (3u << 25)),      // pad_amount:   4 dwords (16B -> LDH=264)
                 (int)(256u << 16),      // tensor_dim0 = 256 (low 16 @ bit48)
                 (int)(256u << 16),      // dim0 hi=0 | tensor_dim1 = 256 (low 16)
                 (int)(256u << 16),      // dim1 hi=0 | tile_dim0 = 256
                 256,                    // tile_dim1 = 256, tile_dim2 = 0
                 256,                    // tensor_dim0_stride = 256 (low 32)
                 0, 0 };
      v4i gz4 = {0, 0, 0, 0};
      v8i gz8 = {0, 0, 0, 0, 0, 0, 0, 0};
      __builtin_amdgcn_tensor_load_to_lds(g0, g1, gz4, gz4, gz8, 0);
      __builtin_amdgcn_s_wait_tensorcnt(0);
    }
    __syncthreads();

    #pragma unroll
    for (int j = 0; j < 4; ++j) {
      const int m = mgrp + 4 * j;
      float xv[8];
      #pragma unroll
      for (int r = 0; r < 8; ++r)
        xv[r] = x[(m * 16 + r + 8 * lhi) * TSTEPS + t];

      v8f acc[4];
      #pragma unroll
      for (int g = 0; g < 4; ++g)
        #pragma unroll
        for (int r = 0; r < 8; ++r)
          acc[g][r] = bsv[g] + xv[r] * wxv[g];       // bias + x outer-product term

      #pragma unroll
      for (int kt = 0; kt < 8; ++kt) {
        const int k0 = kt * 32;
        // A fragment: 16x32 bf16, lanes 0-15 rows, K split {0-7,16-23}/{8-15,24-31}
        const __bf16* hr = &sm.h[(m * 16 + (lane & 15)) * LDH + k0 + lhi * 8];
        v8bf alo = *(const v8bf*)(hr);
        v8bf ahi = *(const v8bf*)(hr + 16);
        v16bf a = __builtin_shufflevector(alo, ahi,
                    0,1,2,3,4,5,6,7,8,9,10,11,12,13,14,15);
        #pragma unroll
        for (int g = 0; g < 4; ++g) {
          // B fragment: 32x16 bf16, lane = column, K split {0-7,16-23}/{8-15,24-31}
          const __bf16* wr = &sm.w[g][ntile * 16 + (lane & 15)][k0 + lhi * 8];
          v8bf blo = *(const v8bf*)(wr);
          v8bf bhi = *(const v8bf*)(wr + 16);
          v16bf bw = __builtin_shufflevector(blo, bhi,
                       0,1,2,3,4,5,6,7,8,9,10,11,12,13,14,15);
          acc[g] = __builtin_amdgcn_wmma_f32_16x16x32_bf16(
                     false, a, false, bw, (short)0, acc[g], false, false);
        }
      }

      // gate nonlinearity + cell update entirely in registers
      #pragma unroll
      for (int r = 0; r < 8; ++r) {
        float gg = fast_tanh(acc[0][r]);
        float ii = fast_sig (acc[1][r]);
        float ff = fast_sig (acc[2][r]);
        float oo = fast_sig (acc[3][r]);
        float c  = gg * ii + cst[j][r] * ff;
        cst[j][r] = c;
        float hh = fast_tanh(c * oo);               // reference: tanh AFTER o-gate
        int b = m * 16 + r + 8 * lhi;
        hg[b * HID + colg] = to_bf16(hh);
        if (t == TSTEPS - 1) h32[b * HID + colg] = hh;
      }
    }
    grid_barrier(cnt, ++bar * NWG);
  }

  // ---- final projection h @ W_ph + b_p (tiny; one WG, one thread per batch row) ----
  if (blockIdx.x == 0) {
    __threadfence();
    int b = tid;
    float accp[NCLS];
    #pragma unroll
    for (int n = 0; n < NCLS; ++n) accp[n] = bp[n];
    for (int k = 0; k < HID; ++k) {
      float hv = h32[b * HID + k];
      #pragma unroll
      for (int n = 0; n < NCLS; ++n) accp[n] += hv * wph[k * NCLS + n];
    }
    #pragma unroll
    for (int n = 0; n < NCLS; ++n) out[b * NCLS + n] = accp[n];
  }
}

extern "C" void kernel_launch(void* const* d_in, const int* in_sizes, int n_in,
                              void* d_out, int out_size, void* d_ws, size_t ws_size,
                              hipStream_t stream) {
  const float* x   = (const float*)d_in[0];
  const float* wgx = (const float*)d_in[1];
  const float* wgh = (const float*)d_in[2];
  const float* bg  = (const float*)d_in[3];
  const float* wix = (const float*)d_in[4];
  const float* wih = (const float*)d_in[5];
  const float* bi  = (const float*)d_in[6];
  const float* wfx = (const float*)d_in[7];
  const float* wfh = (const float*)d_in[8];
  const float* bfv = (const float*)d_in[9];
  const float* wox = (const float*)d_in[10];
  const float* woh = (const float*)d_in[11];
  const float* bo  = (const float*)d_in[12];
  const float* wph = (const float*)d_in[13];
  const float* bp  = (const float*)d_in[14];
  const float* hin = (const float*)d_in[15];

  __bf16*   hg  = (__bf16*)d_ws;                                  // 256*256*2  = 128 KB
  float*    h32 = (float*)((char*)d_ws + 131072);                 // 256*256*4  = 256 KB
  unsigned* cnt = (unsigned*)((char*)d_ws + 131072 + 262144);     // barrier counter
  float*    out = (float*)d_out;

  (void)hipMemsetAsync(cnt, 0, sizeof(unsigned), stream);         // graph-capture legal
  lstm_persistent<<<dim3(NWG), dim3(THREADS), 0, stream>>>(
      x, wgx, wgh, bg, wix, wih, bi, wfx, wfh, bfv,
      wox, woh, bo, wph, bp, hin, hg, h32, cnt, out);
}
